// MetaGL_62491774157182
// MI455X (gfx1250) — matrix operand order repaired and checked by reading, not compiled, not run on hardware
//
#include <hip/hip_runtime.h>
#include <stdint.h>

typedef __attribute__((ext_vector_type(16))) __bf16 bf16x16;
typedef __attribute__((ext_vector_type(8)))  float  v8f;

#define DDIM   256
#define K_TOP  30
#define NSPLIT 4
#define WAVES  8
#define NEG_INF (-1e30f)

static __device__ __forceinline__ unsigned short f2bf_rne(float f) {
    union { float f; unsigned u; } a; a.f = f;
    unsigned u = a.u;
    u += 0x7FFFu + ((u >> 16) & 1u);           // round-to-nearest-even
    return (unsigned short)(u >> 16);
}
static __device__ __forceinline__ float bf2f(unsigned short h) {
    union { unsigned u; float f; } a; a.u = ((unsigned)h) << 16;
    return a.f;
}

// ---------------------------------------------------------------------------
// Kernel 1: row-normalize X and Y, emit bf16 hi/lo split (x ~= hi + lo).
// One wave per row: 32 lanes x 8 floats = 256.
// ---------------------------------------------------------------------------
__global__ void normalize_split_kernel(const float* __restrict__ X,
                                       const float* __restrict__ Y,
                                       unsigned short* __restrict__ Xh,
                                       unsigned short* __restrict__ Xl,
                                       unsigned short* __restrict__ Yh,
                                       unsigned short* __restrict__ Yl,
                                       int Nx, int Ny) {
    const int lane = threadIdx.x & 31;
    const int w    = threadIdx.x >> 5;
    const int r    = blockIdx.x * WAVES + w;
    const float* src; unsigned short* dh; unsigned short* dl; int row;
    if (r < Nx)            { src = X; dh = Xh; dl = Xl; row = r; }
    else if (r < Nx + Ny)  { src = Y; dh = Yh; dl = Yl; row = r - Nx; }
    else return;

    const float4* p = (const float4*)(src + (size_t)row * DDIM + lane * 8);
    float4 a = p[0], b = p[1];
    float ss = a.x*a.x + a.y*a.y + a.z*a.z + a.w*a.w
             + b.x*b.x + b.y*b.y + b.z*b.z + b.w*b.w;
    #pragma unroll
    for (int off = 16; off >= 1; off >>= 1) ss += __shfl_xor(ss, off);
    const float nrm = __builtin_sqrtf(ss);
    const float s   = 1.0f / fmaxf(nrm, 1e-8f);

    float e[8] = { a.x*s, a.y*s, a.z*s, a.w*s, b.x*s, b.y*s, b.z*s, b.w*s };
    unsigned short hh[8], ll[8];
    #pragma unroll
    for (int j = 0; j < 8; ++j) {
        hh[j] = f2bf_rne(e[j]);
        ll[j] = f2bf_rne(e[j] - bf2f(hh[j]));
    }
    union { unsigned short s[8]; uint4 q; } uh, ul;
    #pragma unroll
    for (int j = 0; j < 8; ++j) { uh.s[j] = hh[j]; ul.s[j] = ll[j]; }
    *(uint4*)(dh + (size_t)row * DDIM + lane * 8) = uh.q;
    *(uint4*)(dl + (size_t)row * DDIM + lane * 8) = ul.q;
}

// ---------------------------------------------------------------------------
// Kernel 2: fused GEMM (split-bf16 WMMA) + per-row running top-K.
// Wave w owns 16 rows (A panel resident in VGPRs, hi+lo), sweeps a column
// split in 16-wide tiles. Three INDEPENDENT accumulator chains (hi.hi,
// hi.lo, lo.hi) keep the XDL pipe busy; summed in the epilogue.
// C-tile layout: lane l holds column (l&15); VGPR v holds row v + 8*(l>>4).
// ---------------------------------------------------------------------------
__global__ void __launch_bounds__(256, 1)
gemm_topk_kernel(const unsigned short* __restrict__ Xh,
                 const unsigned short* __restrict__ Xl,
                 const unsigned short* __restrict__ Yh,
                 const unsigned short* __restrict__ Yl,
                 float* __restrict__ pvals, int* __restrict__ pidx,
                 int Nx, int Ny) {
    __shared__ volatile float s_vals[WAVES][16][K_TOP];
    __shared__ volatile int   s_idx [WAVES][16][K_TOP];
    __shared__ volatile float s_thr [WAVES][16];
    __shared__ volatile int   s_mp  [WAVES][16];

    const int lane  = threadIdx.x & 31;
    const int w     = threadIdx.x >> 5;
    const int half  = lane >> 4;
    const int l16   = lane & 15;
    const int rowg0 = blockIdx.x * (WAVES * 16) + w * 16;
    const int split = blockIdx.y;
    const int ncols = Ny / NSPLIT;
    const int nb0   = split * ncols;

    // init per-wave top-K state (no cross-wave sharing -> no barrier needed)
    #pragma unroll
    for (int rr = 0; rr < 16; ++rr)
        if (lane < K_TOP) { s_vals[w][rr][lane] = NEG_INF; s_idx[w][rr][lane] = 0; }
    if (lane < 16) { s_thr[w][lane] = NEG_INF; s_mp[w][lane] = 0; }

    // resident A fragments: 16 rows x 256 K, hi + lo  (128 VGPRs)
    bf16x16 aH[8], aL[8];
    {
        const size_t rbase = (size_t)(rowg0 + l16) * DDIM;
        #pragma unroll
        for (int kb = 0; kb < 8; ++kb) {
            const int koff = kb * 32 + half * 8;
            union { uint4 q[2]; bf16x16 v; } u;
            u.q[0] = *(const uint4*)(Xh + rbase + koff);
            u.q[1] = *(const uint4*)(Xh + rbase + koff + 16);
            aH[kb] = u.v;
            u.q[0] = *(const uint4*)(Xl + rbase + koff);
            u.q[1] = *(const uint4*)(Xl + rbase + koff + 16);
            aL[kb] = u.v;
        }
    }

    for (int nt = 0; nt < ncols; nt += 16) {
        const int    colg  = nb0 + nt + l16;
        const size_t cbase = (size_t)colg * DDIM;
        v8f accHH = {};   // hi.hi chain
        v8f accHL = {};   // hi.lo chain
        v8f accLH = {};   // lo.hi chain
        #pragma unroll
        for (int kb = 0; kb < 8; ++kb) {
            const int koff = kb * 32 + half * 8;
            union { uint4 q[2]; bf16x16 v; } ub;
            ub.q[0] = *(const uint4*)(Yh + cbase + koff);
            ub.q[1] = *(const uint4*)(Yh + cbase + koff + 16);
            bf16x16 bH = ub.v;
            ub.q[0] = *(const uint4*)(Yl + cbase + koff);
            ub.q[1] = *(const uint4*)(Yl + cbase + koff + 16);
            bf16x16 bL = ub.v;
            // independent chains -> back-to-back WMMAs have no RAW hazard
            accLH = __builtin_amdgcn_wmma_f32_16x16x32_bf16(false, aL[kb], false, bH,
                                                            (short)0, accLH, false, false);
            accHL = __builtin_amdgcn_wmma_f32_16x16x32_bf16(false, aH[kb], false, bL,
                                                            (short)0, accHL, false, false);
            accHH = __builtin_amdgcn_wmma_f32_16x16x32_bf16(false, aH[kb], false, bH,
                                                            (short)0, accHH, false, false);
        }
        if (nt + 16 < ncols) {
            __builtin_prefetch(Yh + cbase + (size_t)16 * DDIM, 0, 1);
            __builtin_prefetch(Yl + cbase + (size_t)16 * DDIM, 0, 1);
        }
        v8f acc = accHH + (accHL + accLH);

        // ---- top-K update: threshold test, rare serialized inserts ----
        const int col = nb0 + nt + l16;
        #pragma unroll
        for (int vv = 0; vv < 8; ++vv) {
            const float cand  = acc[vv];
            const int   myrow = vv + 8 * half;
            unsigned pend = (unsigned)__ballot(cand > s_thr[w][myrow]);
            while (pend) {
                const int b    = __ffs(pend) - 1;
                const int rowb = vv + 8 * (b >> 4);
                if (lane == b) {
                    const int p = s_mp[w][rowb];
                    s_vals[w][rowb][p] = cand;
                    s_idx [w][rowb][p] = col;
                }
                // whole wave recomputes min slot of rowb
                float mv = (lane < K_TOP) ? s_vals[w][rowb][lane] : 1e30f;
                int   mp = lane;
                #pragma unroll
                for (int off = 16; off >= 1; off >>= 1) {
                    const float ov = __shfl_xor(mv, off);
                    const int   op = __shfl_xor(mp, off);
                    if (ov < mv || (ov == mv && op < mp)) { mv = ov; mp = op; }
                }
                if (lane == 0) { s_thr[w][rowb] = mv; s_mp[w][rowb] = mp; }
                pend &= ~(1u << b);
                pend &= (unsigned)__ballot(cand > s_thr[w][myrow]);
            }
        }
    }

    // dump partial top-K: [row][split][K_TOP]
    #pragma unroll
    for (int rr = 0; rr < 16; ++rr) {
        if (lane < K_TOP) {
            const size_t o = ((size_t)(rowg0 + rr) * NSPLIT + split) * K_TOP + lane;
            pvals[o] = s_vals[w][rr][lane];
            pidx [o] = s_idx [w][rr][lane];
        }
    }
}

// ---------------------------------------------------------------------------
// Kernel 3: merge NSPLIT partial lists per row -> top-30 sorted descending.
// One wave per row; 120 candidates -> 4 per lane; 30 argmax rounds.
// ---------------------------------------------------------------------------
__global__ void merge_topk_kernel(const float* __restrict__ pvals,
                                  const int*   __restrict__ pidx,
                                  float* __restrict__ outV,
                                  float* __restrict__ outU,
                                  float* __restrict__ outI,
                                  int Nx) {
    const int lane = threadIdx.x & 31;
    const int w    = threadIdx.x >> 5;
    const int r    = blockIdx.x * WAVES + w;
    if (r >= Nx) return;
    const int NC = NSPLIT * K_TOP;   // 120
    float v4[4]; int i4[4];
    #pragma unroll
    for (int j = 0; j < 4; ++j) {
        const int s = lane * 4 + j;
        if (s < NC) { v4[j] = pvals[(size_t)r * NC + s]; i4[j] = pidx[(size_t)r * NC + s]; }
        else        { v4[j] = NEG_INF; i4[j] = 0; }
    }
    for (int round = 0; round < K_TOP; ++round) {
        float bv = v4[0]; int bs = lane * 4;
        #pragma unroll
        for (int j = 1; j < 4; ++j)
            if (v4[j] > bv) { bv = v4[j]; bs = lane * 4 + j; }
        #pragma unroll
        for (int off = 16; off >= 1; off >>= 1) {
            const float ov = __shfl_xor(bv, off);
            const int   os = __shfl_xor(bs, off);
            if (ov > bv || (ov == bv && os < bs)) { bv = ov; bs = os; }
        }
        const int widx = __shfl(i4[bs & 3], bs >> 2);
        if (lane == 0) {
            outV[(size_t)r * K_TOP + round] = bv;
            outU[(size_t)r * K_TOP + round] = (float)r;
            outI[(size_t)r * K_TOP + round] = (float)widx;
        }
        if ((bs >> 2) == lane) v4[bs & 3] = NEG_INF;
    }
}

// ---------------------------------------------------------------------------
extern "C" void kernel_launch(void* const* d_in, const int* in_sizes, int n_in,
                              void* d_out, int out_size, void* d_ws, size_t ws_size,
                              hipStream_t stream) {
    (void)n_in; (void)out_size; (void)ws_size;
    const float* X = (const float*)d_in[0];
    const float* Y = (const float*)d_in[1];
    const int Nx = in_sizes[0] / DDIM;    // 12288
    const int Ny = in_sizes[1] / DDIM;    // 12288

    const size_t nex = (size_t)Nx * DDIM;
    const size_t ney = (size_t)Ny * DDIM;
    unsigned short* Xh = (unsigned short*)d_ws;
    unsigned short* Xl = Xh + nex;
    unsigned short* Yh = Xl + nex;
    unsigned short* Yl = Yh + ney;
    float* pvals = (float*)(Yl + ney);
    int*   pidx  = (int*)(pvals + (size_t)Nx * NSPLIT * K_TOP);

    float* outV = (float*)d_out;
    float* outU = outV + (size_t)Nx * K_TOP;
    float* outI = outU + (size_t)Nx * K_TOP;

    const int totalRows = Nx + Ny;
    normalize_split_kernel<<<(totalRows + WAVES - 1) / WAVES, 256, 0, stream>>>(
        X, Y, Xh, Xl, Yh, Yl, Nx, Ny);

    dim3 g2(Nx / (WAVES * 16), NSPLIT);
    gemm_topk_kernel<<<g2, 256, 0, stream>>>(Xh, Xl, Yh, Yl, pvals, pidx, Nx, Ny);

    merge_topk_kernel<<<(Nx + WAVES - 1) / WAVES, 256, 0, stream>>>(
        pvals, pidx, outV, outU, outI, Nx);
}
